// TrajectoryAgent_19739669692496
// MI455X (gfx1250) — matrix operand order repaired and checked by reading, not compiled, not run on hardware
//
#include <hip/hip_runtime.h>

// ---------------------------------------------------------------------------
// Problem constants (from reference): B=128, E=262144, D=H=128, A=3H=384.
// ---------------------------------------------------------------------------
#define BATCH    128
#define NEDGE    262144
#define HDIM     128
#define ADIM     384

typedef __attribute__((ext_vector_type(16))) __bf16 v16bf;
typedef __attribute__((ext_vector_type(8)))  float  v8f;

// workspace layout (bytes)
#define OFF_Q   0                        // query      [128*384] f32
#define OFF_V   196608                   // V=q@W2^T   [128*384] f32
#define OFF_C   393216                   // c=q.b2     [128]     f32
#define OFF_BP  393728                   // Bpacked    [384*384] bf16 (WMMA layout)

// main-kernel LDS layout (bytes)
#define A_LDS_BYTES 24576                // 2 mtiles * 12 ktiles * 32 lanes * 32B
#define SACC_OFF    24576                // 32 * f32
#define B_LDS_OFF   24704
#define B_LDS_BYTES 294912               // 384*384 bf16
#define SMEM_MAIN   (B_LDS_OFF + B_LDS_BYTES)   // 319616 <= 320KB

__device__ __forceinline__ unsigned short f2bf(float f) {
    __bf16 h = (__bf16)f;                       // RNE f32->bf16
    return __builtin_bit_cast(unsigned short, h);
}

__device__ __forceinline__ v16bf load_v16(const char* p) {
    union { uint4 q[2]; v16bf v; } u;
    u.q[0] = *(const uint4*)p;                  // ds_load_b128
    u.q[1] = *(const uint4*)(p + 16);           // ds_load_b128
    return u.v;
}

// ---------------------------------------------------------------------------
// Prep kernels (tiny: ~88 MFLOP total, negligible vs 77 GF main GEMM)
// ---------------------------------------------------------------------------
// query[b,a] = sum_h hidden[b,h] * Wq[h,a]
__global__ void prep_query(const float* __restrict__ hidden,
                           const float* __restrict__ Wq,
                           float* __restrict__ q) {
    int idx = blockIdx.x * 256 + threadIdx.x;      // 128*384 threads
    int b = idx / ADIM, a = idx % ADIM;
    float s = 0.f;
    for (int h = 0; h < HDIM; ++h)
        s += hidden[b * HDIM + h] * Wq[h * ADIM + a];
    q[idx] = s;
}

// V[b,k] = sum_j W2[k,j] * query[b,j]   (so scores use g . V[b] instead of a 384x384 GEMM per edge)
__global__ void prep_V(const float* __restrict__ q,
                       const float* __restrict__ W2,
                       float* __restrict__ V) {
    int idx = blockIdx.x * 256 + threadIdx.x;      // 128*384 threads
    int b = idx / ADIM, k = idx % ADIM;
    float s = 0.f;
    for (int j = 0; j < ADIM; ++j)
        s += W2[k * ADIM + j] * q[b * ADIM + j];
    V[idx] = s;
}

// c[b] = sum_j b2[j] * query[b,j]
__global__ void prep_c(const float* __restrict__ q,
                       const float* __restrict__ b2,
                       float* __restrict__ c) {
    int b = threadIdx.x;                            // 128 threads
    float s = 0.f;
    for (int j = 0; j < ADIM; ++j)
        s += b2[j] * q[b * ADIM + j];
    c[b] = s;
}

// Ufull[k,n] = sum_h Wsel[k%128,h] * W1[(k/128)*128+h, n], stored in bf16 in the
// 32x16 WMMA-B lane layout used here so the main kernel's LDS image is a plain
// linear copy and every lane's operand is one contiguous 32B chunk.
// B packing used here (32x16, contiguous k): lane = (n%16) | (khalf<<4); within lane, element i
// (0..15) holds K = khalf*16 + i of the 32-K tile.
__global__ void prep_pack(const float* __restrict__ Wr,
                          const float* __restrict__ Wn,
                          const float* __restrict__ Wh,
                          const float* __restrict__ W1,
                          unsigned short* __restrict__ bp) {
    int idx = blockIdx.x * 256 + threadIdx.x;       // 384*384 threads
    int k = idx / ADIM, n = idx % ADIM;
    const float* Ws = (k < 128) ? Wr : (k < 256) ? Wn : Wh;
    int kr = k & 127;
    float s = 0.f;
    for (int h = 0; h < HDIM; ++h)
        s += Ws[kr * HDIM + h] * W1[((k >> 7) * 128 + h) * ADIM + n];
    int kt = k >> 5, kp = k & 31;
    int khalf = kp >> 4, i = kp & 15;
    int lane = (n & 15) | (khalf << 4);
    int blk  = kt * 24 + (n >> 4);
    bp[(blk * 32 + lane) * 16 + i] = f2bf(s);
}

// ---------------------------------------------------------------------------
// Main kernel: per 32-edge tile compute h1 = Acat @ Ufull (+b1), GELU, then
// scores[e] = g_e . V[batch[e]] + c[batch[e]].
// 8 waves = 2(M) x 4(N); each wave: 16 edges x 96 cols, K=384 -> 72 WMMAs.
// ---------------------------------------------------------------------------
__global__ __launch_bounds__(256, 1)
void edge_scores_kernel(const float* __restrict__ rel,
                        const float* __restrict__ node,
                        const float* __restrict__ head,
                        const int*   __restrict__ ebatch,
                        const float* __restrict__ V,
                        const float* __restrict__ cvec,
                        const float* __restrict__ b1,
                        const unsigned short* __restrict__ bpacked,
                        float* __restrict__ out) {
    extern __shared__ __align__(16) char smem[];
    const int t     = threadIdx.x;
    const int lane  = t & 31;
    const int wave  = t >> 5;
    const int waveM = wave >> 2;                 // 0..1 : which 16-edge subtile
    const int waveN = wave & 3;                  // 0..3 : which 96-col slab
    const long long wgBase = (long long)blockIdx.x * 32;

    if (t < 32) *((float*)(smem + SACC_OFF) + t) = 0.0f;

    // --- async DMA of the packed 384x384 bf16 weight image into LDS (288KB) ---
    {
        unsigned           ldsOff = (unsigned)B_LDS_OFF + (unsigned)t * 16u;
        unsigned long long ga     = (unsigned long long)(size_t)bpacked
                                  + (unsigned long long)t * 16ull;
        for (int i = 0; i < 72; ++i) {           // 72 * 256 * 16B = 294912B
            asm volatile("global_load_async_to_lds_b128 %0, %1, off"
                         :: "v"(ldsOff), "v"(ga) : "memory");
            ldsOff += 4096u;
            ga     += 4096ull;
        }
    }

    // --- stage A: 32 edges x 384 K, f32 -> bf16, packed WMMA-A layout ---
    // A packing used here (16x32, contiguous k): lane = (row%16)|(khalf<<4); element i:
    //   K = 16*khalf + i
    for (int it = 0; it < 12; ++it) {
        int u   = it * 256 + t;                  // 0..3071 units of 4 floats
        int row = u / 96;
        int k   = (u % 96) * 4;
        const float* src; int col;
        if (k < 128)      { src = rel;  col = k; }
        else if (k < 256) { src = node; col = k - 128; }
        else              { src = head; col = k - 256; }
        float4 f = *(const float4*)(src + (wgBase + row) * HDIM + col);
        unsigned lo = (unsigned)f2bf(f.x) | ((unsigned)f2bf(f.y) << 16);
        unsigned hi = (unsigned)f2bf(f.z) | ((unsigned)f2bf(f.w) << 16);
        int kp    = k & 31;
        int khalf = kp >> 4;
        int i0    = kp & 15;
        int alane = (row & 15) | (khalf << 4);
        int blk   = (row >> 4) * 12 + (k >> 5);
        *(uint2*)(smem + (blk * 32 + alane) * 32 + i0 * 2) = make_uint2(lo, hi);
    }

    asm volatile("s_wait_asynccnt 0" ::: "memory");
    __syncthreads();

    // --- K loop: 12 x (1 A ds-read, 6 B ds-reads, 6 WMMAs) ---
    v8f zero = {0.f,0.f,0.f,0.f,0.f,0.f,0.f,0.f};
    v8f acc[6];
#pragma unroll
    for (int nt = 0; nt < 6; ++nt) acc[nt] = zero;

    const char* aBase = smem + waveM * 12 * 1024 + lane * 32;
    const char* bBase = smem + B_LDS_OFF + lane * 32;
    for (int kt = 0; kt < 12; ++kt) {
        v16bf a = load_v16(aBase + kt * 1024);
#pragma unroll
        for (int nt = 0; nt < 6; ++nt) {
            int blk = kt * 24 + (waveN * 6 + nt);
            v16bf b = load_v16(bBase + blk * 1024);
            acc[nt] = __builtin_amdgcn_wmma_f32_16x16x32_bf16(
                false, a, false, b, (short)0, acc[nt], false, false);
        }
    }

    // --- epilogue: +b1, exact GELU, dot with V[batch], reduce ---
    // C/D layout: lane holds N=(lane&15); VGPR r holds M = r (lanes<16) or 8+r.
    const int nLocal = lane & 15;
    const int half   = lane >> 4;
    int bidx[8];
#pragma unroll
    for (int r = 0; r < 8; ++r)
        bidx[r] = ebatch[wgBase + waveM * 16 + half * 8 + r];

    float psum[8];
#pragma unroll
    for (int r = 0; r < 8; ++r) psum[r] = 0.f;

#pragma unroll
    for (int nt = 0; nt < 6; ++nt) {
        int col = waveN * 96 + nt * 16 + nLocal;
        float bias = b1[col];
#pragma unroll
        for (int r = 0; r < 8; ++r) {
            float x = acc[nt][r] + bias;
            float g = 0.5f * x * (1.0f + erff(x * 0.70710678118654752f));
            psum[r] += g * V[(long long)bidx[r] * ADIM + col];
        }
    }

    float* sacc = (float*)(smem + SACC_OFF);
#pragma unroll
    for (int r = 0; r < 8; ++r) {
        float s = psum[r];
        s += __shfl_xor(s, 1, 32);
        s += __shfl_xor(s, 2, 32);
        s += __shfl_xor(s, 4, 32);
        s += __shfl_xor(s, 8, 32);               // reduce within each 16-lane half
        if (nLocal == 0)
            atomicAdd(sacc + (waveM * 16 + half * 8 + r), s);  // ds_add_f32
    }
    __syncthreads();

    if (t < 32) {
        long long e = wgBase + t;
        out[e] = sacc[t] + cvec[ebatch[e]];
    }
}

// ---------------------------------------------------------------------------
extern "C" void kernel_launch(void* const* d_in, const int* in_sizes, int n_in,
                              void* d_out, int out_size, void* d_ws, size_t ws_size,
                              hipStream_t stream) {
    const float* hidden = (const float*)d_in[0];
    const float* rel    = (const float*)d_in[1];
    const float* node   = (const float*)d_in[2];
    const float* head   = (const float*)d_in[3];
    const int*   ebatch = (const int*)d_in[4];
    // d_in[5] edge_ptr: unused by reference output
    const float* Wq = (const float*)d_in[6];
    const float* Wr = (const float*)d_in[7];
    const float* Wn = (const float*)d_in[8];
    const float* Wh = (const float*)d_in[9];
    const float* W1 = (const float*)d_in[10];
    const float* b1 = (const float*)d_in[11];
    const float* W2 = (const float*)d_in[12];
    const float* b2 = (const float*)d_in[13];

    char* ws = (char*)d_ws;
    float*          q  = (float*)(ws + OFF_Q);
    float*          V  = (float*)(ws + OFF_V);
    float*          c  = (float*)(ws + OFF_C);
    unsigned short* bp = (unsigned short*)(ws + OFF_BP);

    prep_query<<<(BATCH * ADIM) / 256, 256, 0, stream>>>(hidden, Wq, q);
    prep_V    <<<(BATCH * ADIM) / 256, 256, 0, stream>>>(q, W2, V);
    prep_c    <<<1, BATCH, 0, stream>>>(q, b2, c);
    prep_pack <<<(ADIM * ADIM) / 256, 256, 0, stream>>>(Wr, Wn, Wh, W1, bp);

    hipFuncSetAttribute((const void*)edge_scores_kernel,
                        hipFuncAttributeMaxDynamicSharedMemorySize, SMEM_MAIN);
    edge_scores_kernel<<<NEDGE / 32, 256, SMEM_MAIN, stream>>>(
        rel, node, head, ebatch, V, c, b1, bp, (float*)d_out);
}
